// NaryTree_72550587564073
// MI455X (gfx1250) — matrix operand order repaired and checked by reading, not compiled, not run on hardware
//
#include <hip/hip_runtime.h>

// ---------------------------------------------------------------------------
// Problem constants (from reference): B=8, S=128, H=512, E=512, NSTEPS=8
// Fused GEMM shape used everywhere: M=1024 (B*S), N=2048, K=512.
//
// Matrix math: split-bf16 ("3xBF16") scheme on V_WMMA_F32_16X16X32_BF16.
//   a = a_hi + a_lo (bf16 RNE split); a*b ~= ah*bh + ah*bl + al*bh (f32 acc)
// -> near-fp32 accuracy at the bf16 matrix-pipe rate: 3 WMMAs per K=32 chunk
//    instead of 8 f32 16x16x4 WMMAs (~2.7x fewer matrix instructions).
// ---------------------------------------------------------------------------
#define BB      8
#define SS      128
#define HH      512
#define EE      512
#define NSTEPS  8
#define GM      1024      // B*S
#define GN      2048      // fused N
#define GK      512

typedef float v8f __attribute__((ext_vector_type(8)));
typedef __bf16 v16bf __attribute__((ext_vector_type(16)));
typedef __bf16 v8bf  __attribute__((ext_vector_type(8)));
typedef unsigned short ushort_t;

__device__ __forceinline__ float sigmoidf(float x) {
    return 1.0f / (1.0f + __expf(-x));
}

__device__ __forceinline__ ushort_t bf16_rne(float f) {
    unsigned int u = __float_as_uint(f);
    u += 0x7FFFu + ((u >> 16) & 1u);
    return (ushort_t)(u >> 16);
}
__device__ __forceinline__ float bf16_to_f32(ushort_t h) {
    return __uint_as_float(((unsigned int)h) << 16);
}
__device__ __forceinline__ void bf16_split(float f, ushort_t* hi, ushort_t* lo) {
    ushort_t h = bf16_rne(f);
    *hi = h;
    *lo = bf16_rne(f - bf16_to_f32(h));
}

// ---------------------------------------------------------------------------
// Generic zero-fill
// ---------------------------------------------------------------------------
__global__ __launch_bounds__(256) void zero_kernel(float* __restrict__ p, int n) {
    int i = blockIdx.x * 256 + threadIdx.x;
    if (i < n) p[i] = 0.0f;
}

// ---------------------------------------------------------------------------
// f32 -> bf16 hi/lo split (used for h each step)
// ---------------------------------------------------------------------------
__global__ __launch_bounds__(256) void split_kernel(const float* __restrict__ src,
                                                    ushort_t* __restrict__ hi,
                                                    ushort_t* __restrict__ lo, int n) {
    int i = blockIdx.x * 256 + threadIdx.x;
    if (i < n) {
        ushort_t h, l;
        bf16_split(src[i], &h, &l);
        hi[i] = h; lo[i] = l;
    }
}

// ---------------------------------------------------------------------------
// Embedding gather + split: X[row,k] = emb[ids[row],k] -> bf16 hi/lo
// ---------------------------------------------------------------------------
__global__ __launch_bounds__(256) void embed_split_kernel(const int* __restrict__ ids,
                                                          const float* __restrict__ emb,
                                                          ushort_t* __restrict__ Xh,
                                                          ushort_t* __restrict__ Xl) {
    int i = blockIdx.x * 256 + threadIdx.x;   // over 1024*512
    int row = i >> 9;
    int k   = i & 511;
    ushort_t h, l;
    bf16_split(emb[(long)ids[row] * EE + k], &h, &l);
    Xh[i] = h; Xl[i] = l;
}

// ---------------------------------------------------------------------------
// Weight packing: TRANSPOSED + split.  Output layout Wt[n][k] (2048 x 512 bf16).
// Wx cols: [ W_ioux (1536) | W_fx (512) ]
// Wh cols: [ W_iouh_r[:, :512] | W_iouh_l[:, :512] | W_fh0+W_fh1 | W_fh2+W_fh3 ]
// (Only the first H columns of W_iouh_* matter: the reference scatters
//  iou_h*[..., :H] only -> o/u gates never see the recurrent term.)
// ---------------------------------------------------------------------------
__global__ __launch_bounds__(256) void pack_wx_split_kernel(const float* __restrict__ W_ioux,
                                                            const float* __restrict__ W_fx,
                                                            ushort_t* __restrict__ Wh_,
                                                            ushort_t* __restrict__ Wl_) {
    int i = blockIdx.x * 256 + threadIdx.x;   // over 2048*512, i = n*512 + k
    int n = i >> 9;
    int k = i & 511;
    float v = (n < 1536) ? W_ioux[(long)k * 1536 + n]
                         : W_fx[(long)k * 512 + (n - 1536)];
    ushort_t h, l;
    bf16_split(v, &h, &l);
    Wh_[i] = h; Wl_[i] = l;
}

__global__ __launch_bounds__(256) void pack_wh_split_kernel(const float* __restrict__ Wr,
                                                            const float* __restrict__ Wl,
                                                            const float* __restrict__ Wf0,
                                                            const float* __restrict__ Wf1,
                                                            const float* __restrict__ Wf2,
                                                            const float* __restrict__ Wf3,
                                                            ushort_t* __restrict__ Oh,
                                                            ushort_t* __restrict__ Ol) {
    int i = blockIdx.x * 256 + threadIdx.x;   // over 2048*512, i = n*512 + k
    int n = i >> 9;
    int k = i & 511;
    float v;
    if (n < 512)        v = Wr[(long)k * 1536 + n];                 // ld = 3H
    else if (n < 1024)  v = Wl[(long)k * 1536 + (n - 512)];
    else if (n < 1536)  { int c = n - 1024; v = Wf0[(long)k * 512 + c] + Wf1[(long)k * 512 + c]; }
    else                { int c = n - 1536; v = Wf2[(long)k * 512 + c] + Wf3[(long)k * 512 + c]; }
    ushort_t h, l;
    bf16_split(v, &h, &l);
    Oh[i] = h; Ol[i] = l;
}

__global__ __launch_bounds__(256) void pack_bias_kernel(const float* __restrict__ br,
                                                        const float* __restrict__ bl,
                                                        const float* __restrict__ bf0,
                                                        const float* __restrict__ bf1,
                                                        const float* __restrict__ bf2,
                                                        const float* __restrict__ bf3,
                                                        float* __restrict__ bcat) {
    int n = blockIdx.x * 256 + threadIdx.x;   // over 2048
    float v;
    if (n < 512)        v = br[n];
    else if (n < 1024)  v = bl[n - 512];
    else if (n < 1536)  v = bf0[n - 1024] + bf1[n - 1024];
    else                v = bf2[n - 1536] + bf3[n - 1536];
    bcat[n] = v;
}

// ---------------------------------------------------------------------------
// Split-bf16 WMMA GEMM: C(1024x2048) = A(1024x512) * B(512x2048) [+ bias]
//   A given as bf16 hi/lo, row-major [m][k].
//   B given as bf16 hi/lo, TRANSPOSED [n][k] (so fragments are contiguous).
// Block tile 128x64, 8 waves, each wave owns a 16x64 strip (4 accumulators,
// A fragment reused 4x). K consumed in chunks of 32 (one bf16 WMMA K).
// 12 x v_wmma_f32_16x16x32_bf16 per chunk per wave; 16 chunks.
//
// Fragment layouts used here (16-bit; A/B use contiguous K order, not the ISA table order):
//   A 16x32: lane(m, half): 16 contiguous K values K[half*16 .. +16)
//   B 32x16: lane(n, half): 16 contiguous K values K[half*16 .. +16)
//   C 16x16: VGPR g -> row g (lanes 0-15) / g+8 (lanes 16-31), col = lane&15
// ---------------------------------------------------------------------------
__global__ __launch_bounds__(256) void gemm_bf16split_kernel(const ushort_t* __restrict__ Ah,
                                                             const ushort_t* __restrict__ Al,
                                                             const ushort_t* __restrict__ Bth,
                                                             const ushort_t* __restrict__ Btl,
                                                             const float* __restrict__ bias,
                                                             float* __restrict__ C) {
    constexpr int TM = 128, TN = 64, KB = 32;
    constexpr int STR = KB + 16;                    // 48 bf16 = 96 B row stride (16B/32B aligned)
    __shared__ __align__(32) ushort_t As_h[TM * STR];
    __shared__ __align__(32) ushort_t As_l[TM * STR];
    __shared__ __align__(32) ushort_t Bt_h[TN * STR];
    __shared__ __align__(32) ushort_t Bt_l[TN * STR];

    const int blockN = blockIdx.x * TN;
    const int blockM = blockIdx.y * TM;
    const int tid   = threadIdx.x;
    const int wave  = tid >> 5;                     // wave32 (gfx1250)
    const int lane  = tid & 31;
    const int lmod  = lane & 15;
    const int lhalf = lane >> 4;
    const int wrow  = wave * 16;

    v8f acc0 = {}, acc1 = {}, acc2 = {}, acc3 = {};

    for (int kb = 0; kb < GK; kb += KB) {
        __syncthreads();
        // Stage A hi+lo: 2 arrays x 512 16-byte chunks (128 rows x 4 chunks)
        #pragma unroll
        for (int i = 0; i < 4; ++i) {
            int c   = tid + i * 256;
            int arr = c >> 9;
            int idx = c & 511;
            int m   = idx >> 2;
            int c4  = idx & 3;
            const ushort_t* src = arr ? Al : Ah;
            ushort_t*       dst = arr ? As_l : As_h;
            *reinterpret_cast<uint4*>(&dst[m * STR + c4 * 8]) =
                *reinterpret_cast<const uint4*>(&src[(long)(blockM + m) * GK + kb + c4 * 8]);
        }
        // Stage B^T hi+lo: 2 arrays x 256 16-byte chunks (64 rows x 4 chunks)
        #pragma unroll
        for (int i = 0; i < 2; ++i) {
            int c   = tid + i * 256;
            int arr = c >> 8;
            int idx = c & 255;
            int n   = idx >> 2;
            int c4  = idx & 3;
            const ushort_t* src = arr ? Btl : Bth;
            ushort_t*       dst = arr ? Bt_l : Bt_h;
            *reinterpret_cast<uint4*>(&dst[n * STR + c4 * 8]) =
                *reinterpret_cast<const uint4*>(&src[(long)(blockN + n) * GK + kb + c4 * 8]);
        }
        __syncthreads();

        const int m48 = (wrow + lmod) * STR;
        v8bf ah0 = *reinterpret_cast<const v8bf*>(&As_h[m48 + lhalf * 16]);
        v8bf ah1 = *reinterpret_cast<const v8bf*>(&As_h[m48 + lhalf * 16 + 8]);
        v8bf al0 = *reinterpret_cast<const v8bf*>(&As_l[m48 + lhalf * 16]);
        v8bf al1 = *reinterpret_cast<const v8bf*>(&As_l[m48 + lhalf * 16 + 8]);
        v16bf a_hi = __builtin_shufflevector(ah0, ah1, 0,1,2,3,4,5,6,7,8,9,10,11,12,13,14,15);
        v16bf a_lo = __builtin_shufflevector(al0, al1, 0,1,2,3,4,5,6,7,8,9,10,11,12,13,14,15);

        #pragma unroll
        for (int nf = 0; nf < 4; ++nf) {
            const int bofs = (nf * 16 + lmod) * STR + lhalf * 16;
            v16bf b_hi = *reinterpret_cast<const v16bf*>(&Bt_h[bofs]);
            v16bf b_lo = *reinterpret_cast<const v16bf*>(&Bt_l[bofs]);
            v8f* acc = (nf == 0) ? &acc0 : (nf == 1) ? &acc1 : (nf == 2) ? &acc2 : &acc3;
            *acc = __builtin_amdgcn_wmma_f32_16x16x32_bf16(false, a_hi, false, b_hi, (short)0, *acc, false, false);
            *acc = __builtin_amdgcn_wmma_f32_16x16x32_bf16(false, a_hi, false, b_lo, (short)0, *acc, false, false);
            *acc = __builtin_amdgcn_wmma_f32_16x16x32_bf16(false, a_lo, false, b_hi, (short)0, *acc, false, false);
        }
    }

    // Epilogue: C layout (ISA 7.12.2) + optional bias
    const int gm = blockM + wrow + 8 * lhalf;
    const int gn = blockN + lmod;
    const float bv0 = bias ? bias[gn +  0] : 0.0f;
    const float bv1 = bias ? bias[gn + 16] : 0.0f;
    const float bv2 = bias ? bias[gn + 32] : 0.0f;
    const float bv3 = bias ? bias[gn + 48] : 0.0f;
    #pragma unroll
    for (int g = 0; g < 8; ++g) {
        float* crow = C + (long)(gm + g) * GN + gn;
        crow[ 0] = acc0[g] + bv0;
        crow[16] = acc1[g] + bv1;
        crow[32] = acc2[g] + bv2;
        crow[48] = acc3[g] + bv3;
    }
}

// ---------------------------------------------------------------------------
// Per-step scatter/gather:
//   scat_iou[b, idx_r] += P_r ; scat_iou[b, idx_l] += P_l
//   f = sigmoid(fx_x[b, idx_d] + fh_r[b, idx_r] + fh_l[b, idx_l])
//   scat_fc[b, idx_d] += f * c
// P columns: [0,512)=h@Wr+br, [512,1024)=h@Wl+bl, [1024,1536)=fh_r, [1536,2048)=fh_l
// XO columns: [0,1536)=iou_x, [1536,2048)=fx_x
// ---------------------------------------------------------------------------
__global__ __launch_bounds__(256) void step_scatter_kernel(const float* __restrict__ P,
                                                           const float* __restrict__ XO,
                                                           const float* __restrict__ c,
                                                           const int* __restrict__ tree,
                                                           int step,
                                                           float* __restrict__ scat) {
    int i = blockIdx.x * 256 + threadIdx.x;   // over 1024*512
    int row = i >> 9;
    int k   = i & 511;
    int b   = row >> 7;
    int s   = row & 127;
    int base = ((b * NSTEPS + step) * 3) * SS;
    int d = tree[base + s];
    int r = tree[base + SS + s];
    int l = tree[base + 2 * SS + s];
    int rowd = (b << 7) + d;
    int rowr = (b << 7) + r;
    int rowl = (b << 7) + l;

    atomicAdd(&scat[(long)rowr * HH + k], P[(long)row * GN + k]);
    atomicAdd(&scat[(long)rowl * HH + k], P[(long)row * GN + 512 + k]);

    float f = sigmoidf(XO[(long)rowd * GN + 1536 + k]
                     + P[(long)rowr * GN + 1024 + k]
                     + P[(long)rowl * GN + 1536 + k]);
    float fc = f * c[(long)row * HH + k];
    atomicAdd(&scat[(long)(GM + rowd) * HH + k], fc);   // scat_fc region
}

// ---------------------------------------------------------------------------
// Gates + cell update:
//   i = sig(iou_x_i + scat_iou); o = sig(iou_x_o); u = tanh(iou_x_u)
//   c_full = i*u + scat_fc ; h_full = o * tanh(c_full)
// ---------------------------------------------------------------------------
__global__ __launch_bounds__(256) void step_update_kernel(const float* __restrict__ XO,
                                                          const float* __restrict__ scat,
                                                          float* __restrict__ cfull,
                                                          float* __restrict__ hfull) {
    int i = blockIdx.x * 256 + threadIdx.x;   // over 1024*512
    int row = i >> 9;
    int k   = i & 511;
    float ig = sigmoidf(XO[(long)row * GN + k] + scat[(long)row * HH + k]);
    float og = sigmoidf(XO[(long)row * GN + 512 + k]);
    float ug = tanhf(XO[(long)row * GN + 1024 + k]);
    float cf = ig * ug + scat[(long)(GM + row) * HH + k];
    cfull[i] = cf;
    hfull[i] = og * tanhf(cf);
}

// ---------------------------------------------------------------------------
// Masked-scatter ranks: mask[r] = (idx_d != 0); rank = exclusive prefix over
// the flat 1024 rows (global order, crossing batches — matches reference's
// cumsum over the flattened mask). Single block, Hillis–Steele scan.
// ---------------------------------------------------------------------------
__global__ __launch_bounds__(1024) void mask_rank_kernel(const int* __restrict__ tree,
                                                         int step,
                                                         int* __restrict__ mask,
                                                         int* __restrict__ rank) {
    __shared__ int sm[GM];
    int r = threadIdx.x;
    int b = r >> 7;
    int s = r & 127;
    int d = tree[((b * NSTEPS + step) * 3) * SS + s];
    int m = (d != 0) ? 1 : 0;
    mask[r] = m;
    sm[r] = m;
    __syncthreads();
    for (int off = 1; off < GM; off <<= 1) {
        int v = (r >= off) ? sm[r - off] : 0;
        __syncthreads();
        sm[r] += v;
        __syncthreads();
    }
    rank[r] = sm[r] - m;   // exclusive
}

// ---------------------------------------------------------------------------
// Commit: the j-th masked flat row receives flat row j of h_full/c_full
// (PyTorch masked_scatter semantics as implemented by the reference).
// ---------------------------------------------------------------------------
__global__ __launch_bounds__(256) void step_commit_kernel(const int* __restrict__ mask,
                                                          const int* __restrict__ rank,
                                                          const float* __restrict__ hfull,
                                                          const float* __restrict__ cfull,
                                                          float* __restrict__ h,
                                                          float* __restrict__ c) {
    int i = blockIdx.x * 256 + threadIdx.x;   // over 1024*512
    int row = i >> 9;
    int k   = i & 511;
    if (mask[row]) {
        int j = rank[row];
        h[i] = hfull[(long)j * HH + k];
        c[i] = cfull[(long)j * HH + k];
    }
}

// ---------------------------------------------------------------------------
// Host orchestration
// ---------------------------------------------------------------------------
extern "C" void kernel_launch(void* const* d_in, const int* in_sizes, int n_in,
                              void* d_out, int out_size, void* d_ws, size_t ws_size,
                              hipStream_t stream) {
    const int*   input_ids = (const int*)  d_in[0];
    const int*   tree      = (const int*)  d_in[1];
    const float* emb       = (const float*)d_in[2];
    const float* W_ioux    = (const float*)d_in[3];
    const float* W_iouh_r  = (const float*)d_in[4];
    const float* b_iouh_r  = (const float*)d_in[5];
    const float* W_iouh_l  = (const float*)d_in[6];
    const float* b_iouh_l  = (const float*)d_in[7];
    const float* W_fx      = (const float*)d_in[8];
    const float* W_fh0     = (const float*)d_in[9];
    const float* b_fh0     = (const float*)d_in[10];
    const float* W_fh1     = (const float*)d_in[11];
    const float* b_fh1     = (const float*)d_in[12];
    const float* W_fh2     = (const float*)d_in[13];
    const float* b_fh2     = (const float*)d_in[14];
    const float* W_fh3     = (const float*)d_in[15];
    const float* b_fh3     = (const float*)d_in[16];

    const long nBH = (long)GM * HH;    // 524288
    const long nBO = (long)GM * GN;    // 2097152
    const long nWT = (long)GN * GK;    // 1048576 (transposed weight elements)

    // float region
    float* ws    = (float*)d_ws;
    float* XO    = ws;                  // 1024 x 2048  ([iou_x | fx_x])
    float* P     = XO    + nBO;         // 1024 x 2048
    float* scat  = P     + nBO;         // 2 x (1024 x 512): [scat_iou | scat_fc]
    float* cbuf  = scat  + 2 * nBH;     // 1024 x 512
    float* cfull = cbuf  + nBH;         // 1024 x 512
    float* hfull = cfull + nBH;         // 1024 x 512
    float* bcat  = hfull + nBH;         // 2048
    // bf16 (ushort) region — starts 16B-aligned (all prior sizes mult. of 4)
    ushort_t* us    = (ushort_t*)(bcat + GN);
    ushort_t* Wxt_h = us;               // 2048 x 512 (transposed, bf16 hi)
    ushort_t* Wxt_l = Wxt_h + nWT;
    ushort_t* Wht_h = Wxt_l + nWT;
    ushort_t* Wht_l = Wht_h + nWT;
    ushort_t* Xh    = Wht_l + nWT;      // 1024 x 512
    ushort_t* Xl    = Xh + nBH;
    ushort_t* Hh    = Xl + nBH;
    ushort_t* Hl    = Hh + nBH;
    int*      imask = (int*)(Hl + nBH); // 1024
    int*      irank = imask + GM;       // 1024

    float* h = (float*)d_out;           // output h lives in d_out (1024 x 512)

    const int EB = 2048;                // blocks for 1024*512-sized elementwise work
    dim3 gemmGrid(GN / 64, GM / 128);   // (32, 8)

    // One-time prep
    zero_kernel <<<EB, 256, 0, stream>>>(h,    (int)nBH);
    zero_kernel <<<EB, 256, 0, stream>>>(cbuf, (int)nBH);
    pack_wx_split_kernel<<<4096, 256, 0, stream>>>(W_ioux, W_fx, Wxt_h, Wxt_l);
    pack_wh_split_kernel<<<4096, 256, 0, stream>>>(W_iouh_r, W_iouh_l, W_fh0, W_fh1, W_fh2, W_fh3, Wht_h, Wht_l);
    pack_bias_kernel    <<<   8, 256, 0, stream>>>(b_iouh_r, b_iouh_l, b_fh0, b_fh1, b_fh2, b_fh3, bcat);
    embed_split_kernel  <<<EB,   256, 0, stream>>>(input_ids, emb, Xh, Xl);
    gemm_bf16split_kernel<<<gemmGrid, 256, 0, stream>>>(Xh, Xl, Wxt_h, Wxt_l, (const float*)nullptr, XO);

    // 8 serial tree steps
    for (int t = 0; t < NSTEPS; ++t) {
        split_kernel        <<<EB,    256, 0, stream>>>(h, Hh, Hl, (int)nBH);
        gemm_bf16split_kernel<<<gemmGrid, 256, 0, stream>>>(Hh, Hl, Wht_h, Wht_l, bcat, P);
        zero_kernel         <<<2 * EB, 256, 0, stream>>>(scat, (int)(2 * nBH));
        step_scatter_kernel <<<EB,    256, 0, stream>>>(P, XO, cbuf, tree, t, scat);
        step_update_kernel  <<<EB,    256, 0, stream>>>(XO, scat, cfull, hfull);
        mask_rank_kernel    <<<1,    1024, 0, stream>>>(tree, t, imask, irank);
        step_commit_kernel  <<<EB,    256, 0, stream>>>(imask, irank, hfull, cfull, h, cbuf);
    }
}
